// GraphBertEncoder_45294725103881
// MI455X (gfx1250) — matrix operand order repaired and checked by reading, not compiled, not run on hardware
//
#include <hip/hip_runtime.h>

typedef __attribute__((ext_vector_type(16))) __bf16 v16bf;
typedef __attribute__((ext_vector_type(8)))  __bf16 v8bf;
typedef __attribute__((ext_vector_type(8)))  float  v8f;

static __device__ __forceinline__ __bf16 f2bf(float f) {
  union { float f; unsigned u; } v; v.f = f;
  unsigned r = (v.u + 0x7FFFu + ((v.u >> 16) & 1u)) >> 16;   // RNE
  union { unsigned short s; __bf16 b; } o; o.s = (unsigned short)r; return o.b;
}
static __device__ __forceinline__ float bf2f(__bf16 b) {
  union { unsigned short s; __bf16 b; } i; i.b = b;
  union { unsigned u; float f; } o; o.u = ((unsigned)i.s) << 16; return o.f;
}
static __device__ __forceinline__ float gelu_exact(float x) {
  return 0.5f * x * (1.0f + erff(x * 0.70710678118654752f));
}
static __device__ __forceinline__ v16bf comb16(v8bf lo, v8bf hi) {
  return __builtin_shufflevector(lo, hi, 0, 1, 2, 3, 4, 5, 6, 7, 8, 9, 10, 11,
                                 12, 13, 14, 15);
}

// ---------------------------------------------------------------------------
// Templated WMMA GEMM:  C[M x N] = A[M x K](bf16) * Bt[N x K]^T(bf16) + bias
// BM=64 rows/block, BN columns/block, 8 waves, each wave: 1 M-subtile x NT
// N-subtiles of 16x16, K-step = 32 (v_wmma_f32_16x16x32_bf16).
// Double-buffered LDS staging (one barrier per K-step). All VMEM accesses use
// single base pointers bumped by KT so they stay global_load_b128 (FLAT loads
// would pollute DScnt). Next-tile global loads are issued right after the
// barrier and overlap the WMMA chain.
// EPI: 0 = store bf16, 1 = gelu+store bf16,
//      2 = fused (bias + hprev + emb-bias + res) + LayerNorm, store f32+bf16
//          (requires BN==256, gridDim.y==1), optional final h[:,0] scatter.
// GATHER: A rows gathered from x_bf16 via the fixed top-k index pattern.
// ---------------------------------------------------------------------------
template <int BN, int EPI, bool GATHER>
__global__ __launch_bounds__(256) void wmma_gemm(
    const __bf16* A, const __bf16* __restrict__ Bt,
    const float* __restrict__ bias,
    __bf16* __restrict__ outb, float* outf,
    const float* hprev, const float* __restrict__ ebias,
    const float* __restrict__ res_small,
    const float* __restrict__ lng, const float* __restrict__ lnb,
    float* __restrict__ final_out, const __bf16* __restrict__ xg,
    int K, int Ntot, long row0) {
  constexpr int BM = 64, KT = 32, PITCH = KT + 8;
  constexpr int NT = (BN / 16) / 2;
  constexpr int NB = BN / 64;                             // B loads per thread
  constexpr int BUF = (BM + BN) * PITCH;                  // halfs per buffer
  constexpr size_t AB_BYTES = (size_t)2 * BUF * sizeof(__bf16);
  constexpr size_t LN_BYTES = (EPI == 2) ? (size_t)32 * 260 * sizeof(float) : 0;
  constexpr size_t SH_BYTES = AB_BYTES > LN_BYTES ? AB_BYTES : LN_BYTES;
  __shared__ alignas(16) char shraw[SH_BYTES];
  float* csh = (float*)shraw;

  const int tid = threadIdx.x;
  const int wid = tid >> 5, lane = tid & 31;
  const int msub = wid & 3, nbase = (wid >> 2) * NT;
  const int m0 = blockIdx.x * BM;
  const int n0 = blockIdx.y * BN;
  const int fr = lane & 15, hi = lane >> 4;

  v8f zero = {0.f, 0.f, 0.f, 0.f, 0.f, 0.f, 0.f, 0.f};
  v8f acc[NT];
#pragma unroll
  for (int j = 0; j < NT; ++j) acc[j] = zero;

  // Per-thread global A slot: row tid>>2 of the block, 8 halfs at (tid&3)*8.
  const int arow = tid >> 2;
  const int aseg = (tid & 3) * 8;
  const __bf16* aCur;
  if constexpr (GATHER) {
    long rg = row0 + m0 + arow;          // flattened (token, s) row
    long t = rg / 6; int s = (int)(rg % 6);
    long g = t >> 7; int i = (int)(t & 127);
    int jj;
    if (s < 2) jj = i;                   // idx[i,0]=idx[i,1]=i
    else { int c = s - 2; jj = (c >= i) ? (c + 1) : c; }  // 4 smallest != i
    aCur = xg + ((g << 7) + jj) * (long)K + aseg;
  } else {
    aCur = A + (long)(m0 + arow) * K + aseg;
  }
  // Single B base pointer (keeps address-space inference -> global_load).
  const __bf16* bBase = Bt + (size_t)(n0 + (tid >> 2)) * K + aseg;
  const size_t bRowStride = (size_t)64 * K;   // 64 rows between per-thread slots

  // Preload tile 0 into registers.
  v8bf aReg = *(const v8bf*)aCur;
  v8bf bReg[NB];
#pragma unroll
  for (int i = 0; i < NB; ++i)
    bReg[i] = *(const v8bf*)(bBase + i * bRowStride);

  const int nsteps = K / KT;
  for (int step = 0; step < nsteps; ++step) {
    __bf16* As = (__bf16*)shraw + ((step & 1) ? BUF : 0);
    __bf16* Bs = As + BM * PITCH;

    // Stage current tile (registers -> LDS).
    *(v8bf*)&As[arow * PITCH + aseg] = aReg;
#pragma unroll
    for (int i = 0; i < NB; ++i) {
      int brow = (tid >> 2) + i * 64;
      *(v8bf*)&Bs[brow * PITCH + aseg] = bReg[i];
    }
    __syncthreads();   // s_wait_dscnt 0 + barrier: tile visible, prior reads done

    // Kick off global loads for the next tile (simple pointer bumps); they
    // overlap the WMMAs below.
    if (step + 1 < nsteps) {
      aCur += KT;
      bBase += KT;
      aReg = *(const v8bf*)aCur;
#pragma unroll
      for (int i = 0; i < NB; ++i)
        bReg[i] = *(const v8bf*)(bBase + i * bRowStride);
    }

    // A fragment (16x32 bf16): lane m=fr, halves: K = hi*16..+15
    const __bf16* ap = &As[(msub * 16 + fr) * PITCH + hi * 16];
    v8bf a0 = *(const v8bf*)ap;
    v8bf a1 = *(const v8bf*)(ap + 8);
    // B fragments (32x16 bf16): lane n=fr, halves: K = hi*16..+15.
    const __bf16* bp0 = &Bs[(nbase * 16 + fr) * PITCH + hi * 16];
    v8bf b0 = *(const v8bf*)bp0;
    v8bf b1 = *(const v8bf*)(bp0 + 8);
    v16bf afrag = comb16(a0, a1);
#pragma unroll
    for (int j = 0; j < NT; ++j) {
      v16bf bfrag = comb16(b0, b1);
      if (j + 1 < NT) {
        const __bf16* bp = &Bs[((nbase + j + 1) * 16 + fr) * PITCH + hi * 16];
        b0 = *(const v8bf*)bp;
        b1 = *(const v8bf*)(bp + 8);
      }
      acc[j] = __builtin_amdgcn_wmma_f32_16x16x32_bf16(
          false, afrag, false, bfrag, (short)0, acc[j], false, false);
    }
  }

  if constexpr (EPI == 0 || EPI == 1) {
#pragma unroll
    for (int j = 0; j < NT; ++j) {
      int nc = n0 + (nbase + j) * 16 + fr;
      float bb = bias[nc];
#pragma unroll
      for (int r = 0; r < 8; ++r) {
        int row = m0 + msub * 16 + r + 8 * hi;   // C layout: VGPR r -> M=r+8*hi
        float v = acc[j][r] + bb;
        if constexpr (EPI == 1) v = gelu_exact(v);
        outb[(size_t)row * Ntot + nc] = f2bf(v);
      }
    }
  } else {
    // EPI == 2: fused residual + LayerNorm over full 256-wide row, 2 passes of
    // 32 rows (csh overlays the A/B staging buffers in LDS).
#pragma unroll
    for (int pass = 0; pass < 2; ++pass) {
      __syncthreads();
      if ((msub >> 1) == pass) {
#pragma unroll
        for (int j = 0; j < NT; ++j) {
          int nc = (nbase + j) * 16 + fr;
          float bb = bias[nc];
#pragma unroll
          for (int r = 0; r < 8; ++r) {
            int rloc = msub * 16 + r + 8 * hi;
            long rl = m0 + rloc;
            long rg = row0 + rl;
            float v = acc[j][r] + bb;
            if (hprev) v += hprev[rl * 256 + nc];
            if (ebias) v += ebias[(rg % 6) * 256 + nc];
            if (res_small && rg < 1536) v += res_small[rg * 256 + nc];
            csh[(rloc - pass * 32) * 260 + nc] = v;
          }
        }
      }
      __syncthreads();
      {
        int row = tid >> 3, part = tid & 7;   // 8 threads per row, 32 elems each
        int rloc = pass * 32 + row;
        long rl = m0 + rloc;
        long rg = row0 + rl;
        float s = 0.f, ss = 0.f;
#pragma unroll
        for (int e = 0; e < 32; ++e) {
          float c = csh[row * 260 + part * 32 + e];
          s += c; ss += c * c;
        }
        s += __shfl_xor(s, 1);  s += __shfl_xor(s, 2);  s += __shfl_xor(s, 4);
        ss += __shfl_xor(ss, 1); ss += __shfl_xor(ss, 2); ss += __shfl_xor(ss, 4);
        float mean = s * (1.0f / 256.0f);
        float var = ss * (1.0f / 256.0f) - mean * mean;
        float rstd = rsqrtf(var + 1e-5f);
        bool fin = (final_out != nullptr) && ((rg % 6) == 0);
        float* fo = fin ? (final_out + (rg / 6) * 256) : nullptr;
#pragma unroll
        for (int e = 0; e < 32; ++e) {
          int n = part * 32 + e;
          float c = csh[row * 260 + n];
          float v = (c - mean) * rstd * lng[n] + lnb[n];
          outf[rl * 256 + n] = v;
          outb[rl * 256 + n] = f2bf(v);
          if (fo) fo[n] = v;
        }
      }
    }
  }
}

// ---------------------------------------------------------------------------
// Attention over S=6, H=8, hd=32: one block per token, one wave per head,
// lane = head dim. Scores via wave32 xor-shuffle reductions.
// ---------------------------------------------------------------------------
__global__ __launch_bounds__(256) void attn_kernel(
    const __bf16* __restrict__ qkv, __bf16* __restrict__ o) {
  const int tl = blockIdx.x;
  const int h = threadIdx.x >> 5;
  const int lane = threadIdx.x & 31;
  const float scale = 0.17677669529663687f;  // 1/sqrt(32)
  const __bf16* base = qkv + (size_t)tl * 6 * 768 + h * 32 + lane;
  float q[6], k[6], v[6];
#pragma unroll
  for (int s = 0; s < 6; ++s) {
    q[s] = bf2f(base[s * 768 + 0]);
    k[s] = bf2f(base[s * 768 + 256]);
    v[s] = bf2f(base[s * 768 + 512]);
  }
  __bf16* ob = o + (size_t)tl * 6 * 256 + h * 32 + lane;
#pragma unroll
  for (int sq = 0; sq < 6; ++sq) {
    float sc[6];
#pragma unroll
    for (int sk = 0; sk < 6; ++sk) {
      float p = q[sq] * k[sk];
#pragma unroll
      for (int m = 16; m >= 1; m >>= 1) p += __shfl_xor(p, m);
      sc[sk] = p * scale;
    }
    float mx = sc[0];
#pragma unroll
    for (int sk = 1; sk < 6; ++sk) mx = fmaxf(mx, sc[sk]);
    float den = 0.f;
#pragma unroll
    for (int sk = 0; sk < 6; ++sk) { sc[sk] = __expf(sc[sk] - mx); den += sc[sk]; }
    float inv = 1.0f / den;
    float acc = 0.f;
#pragma unroll
    for (int sk = 0; sk < 6; ++sk) acc += sc[sk] * v[sk];
    ob[sq * 256] = f2bf(acc * inv);
  }
}

// ------------------------- small helper kernels ----------------------------
__global__ void cvt_bf16_kernel(const float* __restrict__ in,
                                __bf16* __restrict__ out, int n) {
  int i = blockIdx.x * 256 + threadIdx.x;
  if (i < n) out[i] = f2bf(in[i]);
}
__global__ void cvt_transpose_kernel(const float* __restrict__ in,
                                     __bf16* __restrict__ out, int K, int N) {
  int i = blockIdx.x * 256 + threadIdx.x;
  if (i < K * N) {
    int k = i / N, n = i % N;
    out[(size_t)n * K + k] = f2bf(in[i]);
  }
}
__global__ void eb_kernel(const float* __restrict__ pos,
                          const float* __restrict__ wl,
                          const float* __restrict__ hop, float* __restrict__ eb) {
  int s = blockIdx.x, n = threadIdx.x;
  int hid = (s < 2) ? 5 : 511;  // dist==1 -> hop 5, dist==0 -> 511
  eb[s * 256 + n] = pos[s * 256 + n] + wl[n] + hop[hid * 256 + n];
}
__global__ void res_kernel(const float* __restrict__ x,
                           const float* __restrict__ W,
                           const float* __restrict__ b, float* __restrict__ res) {
  int r = blockIdx.x, n = threadIdx.x;   // r < 1536
  int g = r / 6, s = r % 6;
  const float* xr = x + (size_t)(g * 128 + s) * 128;
  float acc = b[n];
  for (int kk = 0; kk < 128; ++kk) acc += xr[kk] * W[kk * 256 + n];
  res[(size_t)r * 256 + n] = acc;
}

// ---------------------------------------------------------------------------
extern "C" void kernel_launch(void* const* d_in, const int* in_sizes, int n_in,
                              void* d_out, int out_size, void* d_ws,
                              size_t ws_size, hipStream_t stream) {
  (void)in_sizes; (void)n_in; (void)out_size; (void)ws_size;
  constexpr int T = 32768, F = 128, D = 256, L = 4;
  constexpr int CT = 4096;        // tokens per chunk
  constexpr int RC = CT * 6;      // rows per chunk
  constexpr int NCHUNK = T / CT;  // 8

  const float* x      = (const float*)d_in[0];
  const float* W_raw  = (const float*)d_in[1];
  const float* b_raw  = (const float*)d_in[2];
  const float* wl_emb = (const float*)d_in[3];
  const float* pos_emb= (const float*)d_in[4];
  const float* hop_emb= (const float*)d_in[5];
  const float* ln_g   = (const float*)d_in[6];
  const float* ln_b   = (const float*)d_in[7];
  const float* Wqkv   = (const float*)d_in[8];
  const float* bqkv   = (const float*)d_in[9];
  const float* Wo     = (const float*)d_in[10];
  const float* bo     = (const float*)d_in[11];
  const float* n1g    = (const float*)d_in[12];
  const float* n1b    = (const float*)d_in[13];
  const float* W1     = (const float*)d_in[14];
  const float* b1     = (const float*)d_in[15];
  const float* W2     = (const float*)d_in[16];
  const float* b2     = (const float*)d_in[17];
  const float* n2g    = (const float*)d_in[18];
  const float* n2b    = (const float*)d_in[19];
  const float* W_res  = (const float*)d_in[20];
  const float* b_res  = (const float*)d_in[21];
  float* out = (float*)d_out;

  size_t off = 0;
  auto carve = [&](size_t bytes) -> char* {
    off = (off + 255) & ~(size_t)255;
    char* r = (char*)d_ws + off;
    off += bytes;
    return r;
  };
  __bf16* x_bf  = (__bf16*)carve((size_t)T * F * 2);
  __bf16* wrawT = (__bf16*)carve((size_t)D * F * 2);
  __bf16* wqkvT = (__bf16*)carve((size_t)L * 3 * D * D * 2);
  __bf16* woT   = (__bf16*)carve((size_t)L * D * D * 2);
  __bf16* w1T   = (__bf16*)carve((size_t)L * 4 * D * D * 2);
  __bf16* w2T   = (__bf16*)carve((size_t)L * 4 * D * D * 2);
  float*  ebuf  = (float*)carve(6 * 256 * 4);
  float*  resb  = (float*)carve((size_t)1536 * 256 * 4);
  float*  h_f   = (float*)carve((size_t)RC * 256 * 4);
  __bf16* h_b   = (__bf16*)carve((size_t)RC * 256 * 2);
  __bf16* qkv_b = (__bf16*)carve((size_t)RC * 768 * 2);
  __bf16* o_b   = (__bf16*)carve((size_t)RC * 256 * 2);
  __bf16* ff1_b = (__bf16*)carve((size_t)RC * 1024 * 2);

  // ---- one-time conversions / precompute (re-done every call; deterministic)
  cvt_bf16_kernel<<<(T * F + 255) / 256, 256, 0, stream>>>(x, x_bf, T * F);
  cvt_transpose_kernel<<<(F * D + 255) / 256, 256, 0, stream>>>(W_raw, wrawT, F, D);
  for (int l = 0; l < L; ++l) {
    cvt_transpose_kernel<<<(D * 3 * D + 255) / 256, 256, 0, stream>>>(
        Wqkv + (size_t)l * D * 3 * D, wqkvT + (size_t)l * 3 * D * D, D, 3 * D);
    cvt_transpose_kernel<<<(D * D + 255) / 256, 256, 0, stream>>>(
        Wo + (size_t)l * D * D, woT + (size_t)l * D * D, D, D);
    cvt_transpose_kernel<<<(D * 4 * D + 255) / 256, 256, 0, stream>>>(
        W1 + (size_t)l * D * 4 * D, w1T + (size_t)l * 4 * D * D, D, 4 * D);
    cvt_transpose_kernel<<<(4 * D * D + 255) / 256, 256, 0, stream>>>(
        W2 + (size_t)l * 4 * D * D, w2T + (size_t)l * 4 * D * D, 4 * D, D);
  }
  eb_kernel<<<6, 256, 0, stream>>>(pos_emb, wl_emb, hop_emb, ebuf);
  res_kernel<<<1536, 256, 0, stream>>>(x, W_res, b_res, resb);

  // ---- per-chunk pipeline (tokens are independent) ----
  for (int c = 0; c < NCHUNK; ++c) {
    long row0 = (long)c * CT * 6;
    // embed: gathered x @ W_raw, + b_raw + emb-bias, LN -> h
    wmma_gemm<256, 2, true><<<dim3(RC / 64, 1), 256, 0, stream>>>(
        nullptr, wrawT, b_raw, h_b, h_f, nullptr, ebuf, nullptr, ln_g, ln_b,
        nullptr, x_bf, 128, 256, row0);
    for (int l = 0; l < L; ++l) {
      wmma_gemm<128, 0, false><<<dim3(RC / 64, 6), 256, 0, stream>>>(
          h_b, wqkvT + (size_t)l * 3 * D * D, bqkv + l * 3 * D, qkv_b, nullptr,
          nullptr, nullptr, nullptr, nullptr, nullptr, nullptr, nullptr,
          256, 768, row0);
      attn_kernel<<<CT, 256, 0, stream>>>(qkv_b, o_b);
      // attn out proj + (h + attn + res) LN -> h
      wmma_gemm<256, 2, false><<<dim3(RC / 64, 1), 256, 0, stream>>>(
          o_b, woT + (size_t)l * D * D, bo + l * D, h_b, h_f, h_f, nullptr,
          resb, n1g + l * D, n1b + l * D, nullptr, nullptr, 256, 256, row0);
      // ff1 = gelu(h @ W1 + b1)
      wmma_gemm<128, 1, false><<<dim3(RC / 64, 8), 256, 0, stream>>>(
          h_b, w1T + (size_t)l * 4 * D * D, b1 + l * 4 * D, ff1_b, nullptr,
          nullptr, nullptr, nullptr, nullptr, nullptr, nullptr, nullptr,
          256, 1024, row0);
      // (h + ff1 @ W2 + b2 + res) LN -> h ; last layer scatters h[:,0] to out
      wmma_gemm<256, 2, false><<<dim3(RC / 64, 1), 256, 0, stream>>>(
          ff1_b, w2T + (size_t)l * 4 * D * D, b2 + l * D, h_b, h_f, h_f,
          nullptr, resb, n2g + l * D, n2b + l * D,
          (l == L - 1) ? out : nullptr, nullptr, 1024, 256, row0);
    }
  }
}
